// HSAM_37546604102405
// MI455X (gfx1250) — matrix operand order repaired and checked by reading, not compiled, not run on hardware
//
#include <hip/hip_runtime.h>
#include <math.h>

#define N_NODES 65536
#define N_EDGES 524288
#define N_GRAPHS 128
#define IN_F 64
#define HID 128
#define TN 4   // 16-col tiles per wave (P must be a multiple of 64)

typedef __attribute__((ext_vector_type(16))) __bf16 v16bf;
typedef __attribute__((ext_vector_type(8)))  float  v8f;

#define GEMM_RELU_OUT 2
#define GEMM_SKIP     4
#define GEMM_PACK     8

// ---- order-preserving float<->uint encode for atomicMax on signed floats ----
__device__ __forceinline__ unsigned fenc(float f) {
    unsigned u = __float_as_uint(f);
    return (u & 0x80000000u) ? ~u : (u | 0x80000000u);
}
__device__ __forceinline__ float fdec(unsigned u) {
    unsigned v = (u & 0x80000000u) ? (u & 0x7FFFFFFFu) : ~u;
    return __uint_as_float(v);
}
__device__ __forceinline__ float gelu_f(float v) {
    return 0.5f * v * (1.0f + erff(v * 0.70710678118654752f));
}

// Packed fragment layouts (bf16), 16 elements per lane, 32 lanes per fragment:
//   A:  Apk[ ((tm*(M/32) + kc)*32 + lane)*16 + idx ]
//       lane = 16*half + r ; row = tm*16+r ; k(kc chunk) per ISA 7.12.2 16-bit A
//       element kk (0..31): half=(kk>>3)&1, idx=(kk&7)+(kk>=16?8:0)
//   B:  Bpk[ ((kc*(P/16) + tn)*32 + lane)*16 + idx ]
//       lane = 16*khalf + r ; col = tn*16+r ; k = kc*32 + (i<4 ? 8*khalf+2i : 16+8*khalf+2(i-4)) (+0/1)

// =====================================================================
// WMMA GEMM on pre-packed bf16 fragments. One wave: 16 rows x 64 cols.
// Inner loop: 1 A-fragment load + TN B-fragment loads + TN v_wmma.
// =====================================================================
__global__ void gemm_wmma(const __bf16* __restrict__ Apk, const __bf16* __restrict__ Bpk,
                          const float* __restrict__ bias, float* __restrict__ C,
                          __bf16* __restrict__ Cpk,
                          const float* __restrict__ skipX, const float* __restrict__ skipGate,
                          int N, int M, int P, int flags)
{
    int wave   = blockIdx.x * (blockDim.x >> 5) + (threadIdx.x >> 5);
    int lane   = threadIdx.x & 31;
    int groups = P >> 6;                 // column groups of 64
    int ntiles = N >> 4;
    if (wave >= groups * ntiles) return; // wave-uniform exit
    int tm = wave / groups, tg = wave % groups;
    int half = lane >> 4, r = lane & 15;
    int kT = M >> 5, pT = P >> 4;

    const v16bf* Af = (const v16bf*)Apk;
    const v16bf* Bf = (const v16bf*)Bpk;

    v8f acc[TN];
#pragma unroll
    for (int j = 0; j < TN; ++j) acc[j] = (v8f){};

    for (int kc = 0; kc < kT; ++kc) {
        v16bf a = Af[((size_t)tm * kT + kc) * 32 + lane];
#pragma unroll
        for (int j = 0; j < TN; ++j) {
            v16bf b = Bf[((size_t)kc * pT + (tg * TN + j)) * 32 + lane];
            acc[j] = __builtin_amdgcn_wmma_f32_16x16x32_bf16(
                         false, a, false, b, (short)0, acc[j], false, false);
        }
    }

    float sa = 1.0f, sb = 0.0f;
    if (flags & GEMM_SKIP) {
        float s = 1.0f / (1.0f + __expf(-skipGate[0]));
        sa = s; sb = 1.0f - s;
    }
#pragma unroll
    for (int j = 0; j < TN; ++j) {
        int coln = tg * 64 + 16 * j + r;
        float bia = bias ? bias[coln] : 0.0f;
#pragma unroll
        for (int i = 0; i < 8; ++i) {
            // C/D 16x16 f32 layout: lane -> N=lane%16, VGPR i -> M = i + 8*(lane/16)
            int m = tm * 16 + i + 8 * half;
            float v = acc[j][i] + bia;
            if (flags & GEMM_SKIP) v = sa * v + sb * skipX[(size_t)m * P + coln];
            if (flags & GEMM_RELU_OUT) v = fmaxf(v, 0.0f);
            C[(size_t)m * P + coln] = v;
            if (flags & GEMM_PACK) {
                // re-pack this element as next GEMM's A (width = P)
                int tmn = m >> 4, rn = m & 15;
                int kcn = coln >> 5, kk = coln & 31;
                int hn = (kk >> 3) & 1;
                int idx = (kk & 7) + ((kk & 16) ? 8 : 0);
                Cpk[(((size_t)tmn * (P >> 5) + kcn) * 32 + (hn * 16 + rn)) * 16 + idx] = (__bf16)v;
            }
        }
    }
}

// pack fp32 row-major [N x M] into A-fragment bf16 order
__global__ void pack_a(const float* __restrict__ A, __bf16* __restrict__ Apk, int N, int M)
{
    int t = blockIdx.x * blockDim.x + threadIdx.x;
    int kT = M >> 5;
    int total = (N >> 4) * kT * 32;
    if (t >= total) return;
    int lane = t & 31, rest = t >> 5;
    int kc = rest % kT, tm = rest / kT;
    int half = lane >> 4, r = lane & 15;
    const float2* Arow = (const float2*)(A + (size_t)(tm * 16 + r) * M + kc * 32);
    v16bf a;
#pragma unroll
    for (int i = 0; i < 8; ++i) {
        int kb = (i < 4) ? (8 * half + 2 * i) : (16 + 8 * half + 2 * (i - 4));
        float2 p = Arow[kb >> 1];
        a[2 * i] = (__bf16)p.x; a[2 * i + 1] = (__bf16)p.y;
    }
    ((v16bf*)Apk)[t] = a;
}

// pack fp32 row-major [M x P] weights into B-fragment bf16 order
__global__ void pack_b(const float* __restrict__ B, __bf16* __restrict__ Bpk, int M, int P)
{
    int t = blockIdx.x * blockDim.x + threadIdx.x;
    int pT = P >> 4;
    int total = (M >> 5) * pT * 32;
    if (t >= total) return;
    int lane = t & 31, rest = t >> 5;
    int tn = rest % pT, kc = rest / pT;
    int half = lane >> 4, r = lane & 15;
    int col = tn * 16 + r;
    v16bf b;
#pragma unroll
    for (int i = 0; i < 8; ++i) {
        int k = kc * 32 + ((i < 4) ? (8 * half + 2 * i) : (16 + 8 * half + 2 * (i - 4)));
        b[2 * i]     = (__bf16)B[(size_t)k * P + col];
        b[2 * i + 1] = (__bf16)B[(size_t)(k + 1) * P + col];
    }
    ((v16bf*)Bpk)[t] = b;
}

// fused exact-GELU + A-fragment pack (agg fp32 [N x 128] -> packed bf16)
__global__ void gelu_pack(const float* __restrict__ agg, __bf16* __restrict__ Apk, int N)
{
    size_t i = (size_t)blockIdx.x * blockDim.x + threadIdx.x;
    if (i >= (size_t)N * 128) return;
    int row = (int)(i >> 7), k = (int)(i & 127);
    float v = gelu_f(agg[i]);
    int tm = row >> 4, r = row & 15;
    int kc = k >> 5, kk = k & 31;
    int half = (kk >> 3) & 1;
    int idx = (kk & 7) + ((kk & 16) ? 8 : 0);
    Apk[(((size_t)tm * 4 + kc) * 32 + (half * 16 + r)) * 16 + idx] = (__bf16)v;
}

// =====================================================================
// Fold per-head a_rel / m_rel into Wk / Wv and build fused [fin x 384]
// weight ([Wq | Wk@blockdiag(a_rel) | Wv@blockdiag(m_rel)]) + fused bias.
// =====================================================================
__global__ void prep_wcat(const float* __restrict__ Wk, const float* __restrict__ bk,
                          const float* __restrict__ Wq, const float* __restrict__ bq,
                          const float* __restrict__ Wv, const float* __restrict__ bv,
                          const float* __restrict__ a_rel, const float* __restrict__ m_rel,
                          float* __restrict__ Wcat, float* __restrict__ bcat, int fin)
{
    int idx = blockIdx.x * blockDim.x + threadIdx.x;
    int total = (fin + 1) * 384;
    if (idx >= total) return;
    int rrow = idx / 384, c = idx % 384;
    int sect = c >> 7;            // 0:q  1:k'  2:v'
    int cc = c & 127;
    int h = cc >> 6, e = cc & 63;
    const float *W, *bb, *R = nullptr;
    if (sect == 0)      { W = Wq; bb = bq; }
    else if (sect == 1) { W = Wk; bb = bk; R = a_rel; }
    else                { W = Wv; bb = bv; R = m_rel; }
    float val;
    if (!R) {
        val = (rrow < fin) ? W[rrow * 128 + cc] : bb[cc];
    } else {
        const float* Rh = R + h * 64 * 64;
        float acc = 0.0f;
        if (rrow < fin) { for (int d = 0; d < 64; ++d) acc += W[rrow * 128 + h * 64 + d] * Rh[d * 64 + e]; }
        else            { for (int d = 0; d < 64; ++d) acc += bb[h * 64 + d] * Rh[d * 64 + e]; }
        val = acc;
    }
    if (rrow < fin) Wcat[(size_t)rrow * 384 + c] = val;
    else            bcat[c] = val;
}

__global__ void init_layer(unsigned* __restrict__ nmax, float* __restrict__ nsum,
                           float* __restrict__ agg, int N)
{
    size_t i = (size_t)blockIdx.x * blockDim.x + threadIdx.x;
    if (i < (size_t)2 * N) { nmax[i] = fenc(-3.0e38f); nsum[i] = 0.0f; }
    size_t total = (size_t)N * 128, stride = (size_t)gridDim.x * blockDim.x;
    for (size_t j = i; j < total; j += stride) agg[j] = 0.0f;
}

// One wave per edge: alpha_h = <q[dst,h], k'[src,h]> * p_rel[h] / 8
__global__ void edge_logits(const float* __restrict__ qkv, const int* __restrict__ src,
                            const int* __restrict__ dst, const float* __restrict__ p_rel,
                            float* __restrict__ elog, unsigned* __restrict__ nmax, int E)
{
    int e = blockIdx.x * (blockDim.x >> 5) + (threadIdx.x >> 5);
    int lane = threadIdx.x & 31;
    if (e >= E) return;
    int s = src[e], d = dst[e];
    const float* q = qkv + (size_t)d * 384;
    const float* k = qkv + (size_t)s * 384 + 128;
    float p0 = q[lane] * k[lane] + q[lane + 32] * k[lane + 32];
    float p1 = q[64 + lane] * k[64 + lane] + q[96 + lane] * k[96 + lane];
#pragma unroll
    for (int off = 16; off; off >>= 1) {
        p0 += __shfl_xor(p0, off, 32);
        p1 += __shfl_xor(p1, off, 32);
    }
    if (lane == 0) {
        float l0 = p0 * p_rel[0] * 0.125f;
        float l1 = p1 * p_rel[1] * 0.125f;
        elog[(size_t)e * 2]     = l0;
        elog[(size_t)e * 2 + 1] = l1;
        atomicMax(nmax + (size_t)d * 2,     fenc(l0));
        atomicMax(nmax + (size_t)d * 2 + 1, fenc(l1));
    }
}

__global__ void edge_exp(const int* __restrict__ dst, float* __restrict__ elog,
                         const unsigned* __restrict__ nmax, float* __restrict__ nsum, int E)
{
    int i = blockIdx.x * blockDim.x + threadIdx.x;
    if (i >= 2 * E) return;
    int e = i >> 1, h = i & 1;
    int d = dst[e];
    float ex = __expf(elog[i] - fdec(nmax[(size_t)d * 2 + h]));
    elog[i] = ex;
    atomicAdd(nsum + (size_t)d * 2 + h, ex);
}

// agg[dst] += v'[src] * alpha  (128 threads per edge, f32 global atomics)
__global__ void edge_aggregate(const float* __restrict__ qkv, const int* __restrict__ src,
                               const int* __restrict__ dst, const float* __restrict__ elog,
                               const float* __restrict__ nsum, float* __restrict__ agg, int E)
{
    int e = blockIdx.x * 2 + (threadIdx.x >> 7);
    int f = threadIdx.x & 127;
    if (e >= E) return;
    int s = src[e], d = dst[e];
    int h = f >> 6;
    float a = elog[(size_t)e * 2 + h] / (nsum[(size_t)d * 2 + h] + 1e-16f);
    float v = qkv[(size_t)s * 384 + 256 + f];
    atomicAdd(agg + (size_t)d * 128 + f, v * a);
}

// Node-weight MLP: relu([x,goal]@W1+b1)@W2+b2 ; atomicMax per graph
__global__ void wmlp_k(const float* __restrict__ x, const float* __restrict__ goal,
                       const int* __restrict__ batch,
                       const float* __restrict__ W1, const float* __restrict__ b1,
                       const float* __restrict__ W2, const float* __restrict__ b2,
                       float* __restrict__ wlog, unsigned* __restrict__ gmax, int N)
{
    int n = blockIdx.x * blockDim.x + threadIdx.x;
    if (n >= N) return;
    const float* xr = x + (size_t)n * 64;
    const float* gr = goal + (size_t)n * 10;
    float out = b2[0];
    for (int j = 0; j < 32; ++j) {
        float acc = b1[j];
        for (int i = 0; i < 64; ++i) acc += xr[i] * W1[i * 32 + j];
        for (int i = 0; i < 10; ++i) acc += gr[i] * W1[(64 + i) * 32 + j];
        out += fmaxf(acc, 0.0f) * W2[j];
    }
    wlog[n] = out;
    atomicMax(gmax + batch[n], fenc(out));
}

__global__ void wexp_k(const int* __restrict__ batch, float* __restrict__ wlog,
                       const unsigned* __restrict__ gmax, float* __restrict__ gsum, int N)
{
    int n = blockIdx.x * blockDim.x + threadIdx.x;
    if (n >= N) return;
    int b = batch[n];
    float ex = __expf(wlog[n] - fdec(gmax[b]));
    wlog[n] = ex;
    atomicAdd(gsum + b, ex);
}

__global__ void init_small(unsigned* __restrict__ gmax, float* __restrict__ gsum,
                           float* __restrict__ g)
{
    int i = blockIdx.x * blockDim.x + threadIdx.x;
    if (i < 128) { gmax[i] = fenc(-3.0e38f); gsum[i] = 0.0f; }
    if (i < 128 * 128) g[i] = 0.0f;
}

__global__ void pool_k(const float* __restrict__ h, const int* __restrict__ batch,
                       const float* __restrict__ wexp, const float* __restrict__ gsum,
                       float* __restrict__ g, int N)
{
    size_t i = (size_t)blockIdx.x * blockDim.x + threadIdx.x;
    if (i >= (size_t)N * 128) return;
    int n = (int)(i >> 7), f = (int)(i & 127);
    int b = batch[n];
    float w = wexp[n] / (gsum[b] + 1e-16f);
    atomicAdd(g + (size_t)b * 128 + f, h[i] * w);
}

extern "C" void kernel_launch(void* const* d_in, const int* in_sizes, int n_in,
                              void* d_out, int out_size, void* d_ws, size_t ws_size,
                              hipStream_t stream)
{
    const int N = N_NODES, E = N_EDGES;
    const float* x    = (const float*)d_in[0];
    const float* goal = (const float*)d_in[1];
    const int*   eidx = (const int*)d_in[2];
    const int*   batch= (const int*)d_in[3];
    const int* src = eidx;
    const int* dst = eidx + E;
    const int cbase[3] = {4, 16, 28};
    const float* wW1 = (const float*)d_in[40];
    const float* wb1 = (const float*)d_in[41];
    const float* wW2 = (const float*)d_in[42];
    const float* wb2 = (const float*)d_in[43];
    const float* mW1 = (const float*)d_in[44];
    const float* mb1 = (const float*)d_in[45];
    const float* mW2 = (const float*)d_in[46];
    const float* mb2 = (const float*)d_in[47];

    // ---- workspace carve (units of float; bf16 buffers counted as half) ----
    float* ws = (float*)d_ws;
    size_t off = 0;
    float* qkv  = ws + off; off += (size_t)N * 384;
    float* hA   = ws + off; off += (size_t)N * 128;
    float* hB   = ws + off; off += (size_t)N * 128;
    float* agg  = ws + off; off += (size_t)N * 128;
    float* elog = ws + off; off += (size_t)E * 2;
    unsigned* nmax = (unsigned*)(ws + off); off += (size_t)N * 2;
    float* nsum = ws + off; off += (size_t)N * 2;
    float* Wcat = ws + off; off += (size_t)128 * 384;
    float* bcat = ws + off; off += 384;
    float* wlog = ws + off; off += N;
    unsigned* gmax = (unsigned*)(ws + off); off += 128;
    float* gsum = ws + off; off += 128;
    float* g    = ws + off; off += 128 * 128;
    float* g2   = ws + off; off += 128 * 128;
    __bf16* apk   = (__bf16*)(ws + off); off += (size_t)N * 64;   // N*128 bf16
    __bf16* aggP  = (__bf16*)(ws + off); off += (size_t)N * 64;   // N*128 bf16
    __bf16* WcatP = (__bf16*)(ws + off); off += 128 * 192;        // 128*384 bf16
    __bf16* WaP   = (__bf16*)(ws + off); off += 128 * 64;
    __bf16* mW1P  = (__bf16*)(ws + off); off += 128 * 64;
    __bf16* mW2P  = (__bf16*)(ws + off); off += 128 * 128;
    __bf16* gP    = (__bf16*)(ws + off); off += 128 * 64;
    __bf16* g2P   = (__bf16*)(ws + off); off += 128 * 64;
    (void)ws_size; (void)n_in; (void)in_sizes; (void)out_size;

    // pack layer-0 activations (x) and head-MLP weights
    {
        int t = (N / 16) * (IN_F / 32) * 32;
        pack_a<<<(t + 255) / 256, 256, 0, stream>>>(x, apk, N, IN_F);
    }
    {
        int t = (128 / 32) * (128 / 16) * 32;
        pack_b<<<(t + 255) / 256, 256, 0, stream>>>(mW1, mW1P, 128, 128);
    }
    {
        int t = (128 / 32) * (256 / 16) * 32;
        pack_b<<<(t + 255) / 256, 256, 0, stream>>>(mW2, mW2P, 128, 256);
    }

    const float* Xin = x;   // fp32 view of current activations (for skip)
    int fin = IN_F;
    float* Xout = hA;

    for (int L = 0; L < 3; ++L) {
        const int pb = cbase[L];
        const float* Wk    = (const float*)d_in[pb + 0];
        const float* bk    = (const float*)d_in[pb + 1];
        const float* Wq    = (const float*)d_in[pb + 2];
        const float* bq    = (const float*)d_in[pb + 3];
        const float* Wv    = (const float*)d_in[pb + 4];
        const float* bv    = (const float*)d_in[pb + 5];
        const float* Wa    = (const float*)d_in[pb + 6];
        const float* ba    = (const float*)d_in[pb + 7];
        const float* a_rel = (const float*)d_in[pb + 8];
        const float* m_rel = (const float*)d_in[pb + 9];
        const float* p_rel = (const float*)d_in[pb + 10];
        const float* skipp = (const float*)d_in[pb + 11];

        // fused [fin x 384] weight + pack to fragments
        int ptot = (fin + 1) * 384;
        prep_wcat<<<(ptot + 255) / 256, 256, 0, stream>>>(
            Wk, bk, Wq, bq, Wv, bv, a_rel, m_rel, Wcat, bcat, fin);
        {
            int t = (fin / 32) * (384 / 16) * 32;
            pack_b<<<(t + 255) / 256, 256, 0, stream>>>(Wcat, WcatP, fin, 384);
        }
        {
            int t = (128 / 32) * (128 / 16) * 32;
            pack_b<<<(t + 255) / 256, 256, 0, stream>>>(Wa, WaP, 128, 128);
        }

        // fused q/k'/v' GEMM:  qkv = Xin @ Wcat + bcat   (reads packed A)
        {
            int waves = (N / 16) * (384 / 64);
            gemm_wmma<<<(waves + 3) / 4, 128, 0, stream>>>(
                apk, WcatP, bcat, qkv, nullptr, nullptr, nullptr, N, fin, 384, 0);
        }

        init_layer<<<(N * 128) / 256, 256, 0, stream>>>(nmax, nsum, agg, N);
        edge_logits<<<E / 8, 256, 0, stream>>>(qkv, src, dst, p_rel, elog, nmax, E);
        edge_exp<<<(2 * E) / 256, 256, 0, stream>>>(dst, elog, nmax, nsum, E);
        edge_aggregate<<<E / 2, 256, 0, stream>>>(qkv, src, dst, elog, nsum, agg, E);

        // gelu(agg) -> packed A fragments, then
        // out = relu( gate*(gelu(agg)@Wa + ba) + (1-gate)*Xin ), also re-packed
        gelu_pack<<<(N * 128) / 256, 256, 0, stream>>>(agg, aggP, N);
        {
            int flags = GEMM_RELU_OUT | GEMM_PACK | (L > 0 ? GEMM_SKIP : 0);
            int waves = (N / 16) * (128 / 64);
            gemm_wmma<<<(waves + 3) / 4, 128, 0, stream>>>(
                aggP, WaP, ba, Xout, apk, Xin, skipp, N, 128, 128, flags);
        }

        Xin = Xout;
        fin = HID;
        Xout = (Xout == hA) ? hB : hA;
    }

    // node-weight softmax over graphs + weighted pooling
    init_small<<<64, 256, 0, stream>>>(gmax, gsum, g);
    wmlp_k<<<N / 256, 256, 0, stream>>>(x, goal, batch, wW1, wb1, wW2, wb2, wlog, gmax, N);
    wexp_k<<<N / 256, 256, 0, stream>>>(batch, wlog, gmax, gsum, N);
    pool_k<<<(N * 128) / 256, 256, 0, stream>>>(Xin, batch, wlog, gsum, g, N);

    // head MLP: relu(g@W1+b1) @ W2 + b2 -> d_out [128 x 256]
    {
        int t = (128 / 16) * (128 / 32) * 32;
        pack_a<<<(t + 255) / 256, 256, 0, stream>>>(g, gP, 128, 128);
    }
    {
        int waves = (128 / 16) * (128 / 64);
        gemm_wmma<<<(waves + 3) / 4, 128, 0, stream>>>(
            gP, mW1P, mb1, g2, g2P, nullptr, nullptr, 128, 128, 128,
            GEMM_RELU_OUT | GEMM_PACK);
    }
    {
        int waves = (128 / 16) * (256 / 64);
        gemm_wmma<<<(waves + 3) / 4, 128, 0, stream>>>(
            g2P, mW2P, mb2, (float*)d_out, nullptr, nullptr, nullptr, 128, 128, 256, 0);
    }
}
